// JointCoAttn_68719476736311
// MI455X (gfx1250) — matrix operand order repaired and checked by reading, not compiled, not run on hardware
//
#include <hip/hip_runtime.h>
#include <math.h>

typedef __attribute__((ext_vector_type(16))) _Float16 v16h;
typedef __attribute__((ext_vector_type(8)))  _Float16 v8h;
typedef __attribute__((ext_vector_type(4)))  _Float16 v4h;
typedef __attribute__((ext_vector_type(8)))  float    v8f;

namespace {
constexpr int Bb = 64, Tt = 10, Dd = 1024, Ee = 2048;
constexpr float kScale = 0.022097086912079608f;  // 1/sqrt(2048)
}

__device__ __forceinline__ v16h mk16(v8h lo, v8h hi){
  union U { v16h v; v8h h[2]; } u; u.h[0] = lo; u.h[1] = hi; return u.v;
}
__device__ __forceinline__ v8f zero8f(){
  v8f z = {0.f,0.f,0.f,0.f,0.f,0.f,0.f,0.f}; return z;
}
__device__ __forceinline__ v8h zero8h(){
  v8h z = {(_Float16)0.f,(_Float16)0.f,(_Float16)0.f,(_Float16)0.f,
           (_Float16)0.f,(_Float16)0.f,(_Float16)0.f,(_Float16)0.f};
  return z;
}
__device__ __forceinline__ v8f wmma16(v16h a, v16h b, v8f c){
  return __builtin_amdgcn_wmma_f32_16x16x32_f16(false, a, false, b, (short)0, c, false, false);
}
// hardware V_TANH_F32 if the builtin exists on this toolchain, else libm
__device__ __forceinline__ float fast_tanh(float x){
#if __has_builtin(__builtin_amdgcn_tanhf)
  return __builtin_amdgcn_tanhf(x);
#elif __has_builtin(__builtin_amdgcn_tanh_f32)
  return __builtin_amdgcn_tanh_f32(x);
#else
  return tanhf(x);
#endif
}

// ---------------- LDS-tiled weight transpose + f32->f16:  Wt[n][k] = W[k][n] ----------------
// grid (K/32, N/32), block (32,8). Coalesced reads along n, coalesced writes along k.
__global__ void k_wtrans(const float* __restrict__ W, _Float16* __restrict__ Wt,
                         int K, int N){
  __shared__ _Float16 tile[32][33];
  const int kb = blockIdx.x * 32, nb = blockIdx.y * 32;
  const int tx = threadIdx.x, ty = threadIdx.y;
  #pragma unroll
  for(int j = 0; j < 4; j++){
    tile[ty + j*8][tx] = (_Float16)W[(long)(kb + ty + j*8) * N + (nb + tx)];
  }
  __syncthreads();
  #pragma unroll
  for(int j = 0; j < 4; j++){
    Wt[(long)(nb + ty + j*8) * K + (kb + tx)] = tile[tx][ty + j*8];
  }
}

// ---------- key projection: key[b,d,t] = sum_t' X[b,t',d]*Wk[t',t] + bk[t] ----------
// stored as f16 [B][D][16] with t=10..15 zeroed (zero-padded K for the scores WMMA)
__global__ void k_keyproj(const float* __restrict__ X, const float* __restrict__ Wk,
                          const float* __restrict__ bk, _Float16* __restrict__ keyo){
  __shared__ float sW[Tt * Tt];
  __shared__ float sb[Tt];
  if(threadIdx.x < Tt * Tt) sW[threadIdx.x] = Wk[threadIdx.x];
  if(threadIdx.x < Tt)      sb[threadIdx.x] = bk[threadIdx.x];
  __syncthreads();
  int gid = blockIdx.x * blockDim.x + threadIdx.x;   // over B*D
  int b = gid >> 10, d = gid & (Dd - 1);
  float x[Tt];
  #pragma unroll
  for(int t = 0; t < Tt; t++) x[t] = X[((long)b * Tt + t) * Dd + d];
  __align__(16) _Float16 o[16];
  #pragma unroll
  for(int t = 0; t < Tt; t++){
    float acc = sb[t];
    #pragma unroll
    for(int u = 0; u < Tt; u++) acc += x[u] * sW[u * Tt + t];
    o[t] = (_Float16)acc;
  }
  #pragma unroll
  for(int t = Tt; t < 16; t++) o[t] = (_Float16)0.f;
  _Float16* dst = keyo + (long)gid * 16;
  *(v8h*)dst       = *(v8h*)&o[0];
  *(v8h*)(dst + 8) = *(v8h*)&o[8];
}

// ---------- joint = concat(audio,video) @ Wq + bq, out f16 [B][E][16] (e-major) ----------
// 64-wide K chunks: 4 WMMAs / wave / iter, 2 barriers / iter
__global__ void k_projjoint(const float* __restrict__ audio, const float* __restrict__ video,
                            const _Float16* __restrict__ WT, const float* __restrict__ bias,
                            _Float16* __restrict__ outp){
  __shared__ __align__(16) _Float16 Ach[16][64];
  const int b = blockIdx.y;
  const int n0 = blockIdx.x * 128;
  const int tid = threadIdx.x;
  const int lane = tid & 31, wave = tid >> 5;
  const int l16 = lane & 15, lh = lane >> 4;
  v8f acc0 = zero8f(), acc1 = zero8f();
  const int st = tid >> 3, sko = (tid & 7) * 8;
  const _Float16* brow = WT + (long)(n0 + wave*32 + l16) * Ee + lh*16;
  for(int kb = 0; kb < Ee; kb += 64){
    __syncthreads();
    {
      __align__(16) _Float16 tmp[8];
      if(st < Tt){
        int k = kb + sko;   // 8-run never straddles the 1024 concat boundary
        const float* src = (k < Dd) ? (audio + ((long)b * Tt + st) * Dd + k)
                                    : (video + ((long)b * Tt + st) * Dd + (k - Dd));
        float4 f0 = *(const float4*)src;
        float4 f1 = *(const float4*)(src + 4);
        tmp[0]=(_Float16)f0.x; tmp[1]=(_Float16)f0.y; tmp[2]=(_Float16)f0.z; tmp[3]=(_Float16)f0.w;
        tmp[4]=(_Float16)f1.x; tmp[5]=(_Float16)f1.y; tmp[6]=(_Float16)f1.z; tmp[7]=(_Float16)f1.w;
      } else {
        #pragma unroll
        for(int j = 0; j < 8; j++) tmp[j] = (_Float16)0.f;
      }
      *(v8h*)&Ach[st][sko] = *(v8h*)tmp;
    }
    __syncthreads();
    __builtin_prefetch(brow + kb + 64, 0, 1);
    __builtin_prefetch(brow + (long)16*Ee + kb + 64, 0, 1);
    #pragma unroll
    for(int kc = 0; kc < 2; kc++){
      v16h af = mk16(*(v8h*)&Ach[l16][kc*32 + lh*16], *(v8h*)&Ach[l16][kc*32 + lh*16 + 8]);
      const _Float16* bp  = brow + kb + kc*32;
      const _Float16* bp1 = bp + (long)16 * Ee;
      v16h b0 = mk16(*(const v8h*)bp,  *(const v8h*)(bp  + 8));
      v16h b1 = mk16(*(const v8h*)bp1, *(const v8h*)(bp1 + 8));
      acc0 = wmma16(af, b0, acc0);
      acc1 = wmma16(af, b1, acc1);
    }
  }
  #pragma unroll
  for(int s = 0; s < 2; s++){
    int e = n0 + wave*32 + s*16 + l16;
    float bv = bias[e];
    v8f a = s ? acc1 : acc0;
    __align__(16) _Float16 o[8];
    #pragma unroll
    for(int i = 0; i < 8; i++) o[i] = (_Float16)(a[i] + bv);
    *(v8h*)(outp + ((long)(b * Ee + e)) * 16 + lh*8) = *(v8h*)o;
  }
}

// ---------- val = X @ Wv + bv, out f16 [B][16][D] (t-major rows, t>=10 garbage-but-finite) ----------
__global__ void k_projval(const float* __restrict__ X, const _Float16* __restrict__ WT,
                          const float* __restrict__ bias, _Float16* __restrict__ outp){
  __shared__ __align__(16) _Float16 Ach[16][64];
  const int b = blockIdx.y;
  const int n0 = blockIdx.x * 128;
  const int tid = threadIdx.x;
  const int lane = tid & 31, wave = tid >> 5;
  const int l16 = lane & 15, lh = lane >> 4;
  v8f acc0 = zero8f(), acc1 = zero8f();
  const int st = tid >> 3, sko = (tid & 7) * 8;
  const _Float16* brow = WT + (long)(n0 + wave*32 + l16) * Dd + lh*16;
  for(int kb = 0; kb < Dd; kb += 64){
    __syncthreads();
    {
      __align__(16) _Float16 tmp[8];
      if(st < Tt){
        const float* src = X + ((long)b * Tt + st) * Dd + kb + sko;
        float4 f0 = *(const float4*)src;
        float4 f1 = *(const float4*)(src + 4);
        tmp[0]=(_Float16)f0.x; tmp[1]=(_Float16)f0.y; tmp[2]=(_Float16)f0.z; tmp[3]=(_Float16)f0.w;
        tmp[4]=(_Float16)f1.x; tmp[5]=(_Float16)f1.y; tmp[6]=(_Float16)f1.z; tmp[7]=(_Float16)f1.w;
      } else {
        #pragma unroll
        for(int j = 0; j < 8; j++) tmp[j] = (_Float16)0.f;
      }
      *(v8h*)&Ach[st][sko] = *(v8h*)tmp;
    }
    __syncthreads();
    __builtin_prefetch(brow + kb + 64, 0, 1);
    __builtin_prefetch(brow + (long)16*Dd + kb + 64, 0, 1);
    #pragma unroll
    for(int kc = 0; kc < 2; kc++){
      v16h af = mk16(*(v8h*)&Ach[l16][kc*32 + lh*16], *(v8h*)&Ach[l16][kc*32 + lh*16 + 8]);
      const _Float16* bp  = brow + kb + kc*32;
      const _Float16* bp1 = bp + (long)16 * Dd;
      v16h b0 = mk16(*(const v8h*)bp,  *(const v8h*)(bp  + 8));
      v16h b1 = mk16(*(const v8h*)bp1, *(const v8h*)(bp1 + 8));
      acc0 = wmma16(af, b0, acc0);
      acc1 = wmma16(af, b1, acc1);
    }
  }
  #pragma unroll
  for(int s = 0; s < 2; s++){
    int n = n0 + wave*32 + s*16 + l16;
    float bv = bias[n];
    v8f a = s ? acc1 : acc0;
    #pragma unroll
    for(int i = 0; i < 8; i++){
      int t = i + 8*lh;
      outp[((long)b * 16 + t) * Dd + n] = (_Float16)(a[i] + bv);
    }
  }
}

// ---------- fused attention branch: scores=tanh(scale*key@joint) never hits HBM ----------
// 64-wide d chunks: 8 score WMMAs + 4 weighted WMMAs per wave per iter, 16 iters
__global__ void k_attn(const _Float16* __restrict__ keyp,   // [B][D][16]
                       const _Float16* __restrict__ valp,   // [B][16][D]
                       const _Float16* __restrict__ jointp, // [B][E][16]
                       _Float16* __restrict__ outp){        // [B][E][16]
  __shared__ __align__(16) _Float16 sJ[128][16];  // joint tile (e-major, 16 padded t)
  __shared__ __align__(16) _Float16 sK[64][16];   // key d-chunk
  __shared__ __align__(16) _Float16 sV[16][64];   // val d-chunk
  __shared__ __align__(16) _Float16 sS[128][64];  // tanh'd scores [e][d]
  const int b = blockIdx.y;
  const int e0 = blockIdx.x * 128;
  const int tid = threadIdx.x;
  const int lane = tid & 31, wave = tid >> 5;
  const int l16 = lane & 15, lh = lane >> 4;
  {
    const _Float16* src = jointp + ((long)(b * Ee + e0 + tid)) * 16;
    *(v8h*)&sJ[tid][0] = *(const v8h*)src;
    *(v8h*)&sJ[tid][8] = *(const v8h*)(src + 8);
  }
  __syncthreads();
  const int eL = wave * 32 + l16;
  // B-fragments of joint (K=t); K>=16 lanes multiply against zero-padded key rows
  v16h jf0 = mk16(*(v8h*)&sJ[eL][lh*8],    *(v8h*)&sJ[eL][8]);
  v16h jf1 = mk16(*(v8h*)&sJ[eL+16][lh*8], *(v8h*)&sJ[eL+16][8]);
  v8h zh = zero8h();
  v8f accw0 = zero8f(), accw1 = zero8f();
  const int skr = tid >> 1, skj = (tid & 1) * 8;   // sK stage map
  const int svt = tid >> 3, svj = (tid & 7) * 8;   // sV stage map
  const _Float16* kbase = keyp + (long)b * Dd * 16;
  const _Float16* vbase = valp + (long)b * 16 * Dd;
  for(int c = 0; c < 16; ++c){
    const int d0 = c * 64;
    __syncthreads();
    {
      *(v8h*)&sK[skr][skj] = *(const v8h*)(kbase + (long)(d0 + skr) * 16 + skj);
      *(v8h*)&sV[svt][svj] = *(const v8h*)(vbase + (long)svt * Dd + d0 + svj);
      __builtin_prefetch(kbase + (long)(d0 + 64 + skr) * 16 + skj, 0, 1);
      __builtin_prefetch(vbase + (long)svt * Dd + d0 + 64 + svj, 0, 1);
    }
    __syncthreads();
    #pragma unroll
    for(int m = 0; m < 4; ++m){     // four 16-row d sub-tiles
      v16h kf = mk16(*(v8h*)&sK[m*16 + l16][lh*8], zh);   // K(t)=16..31 zero
      v8f s0 = wmma16(kf, jf0, zero8f());
      v8f s1 = wmma16(kf, jf1, zero8f());
      __align__(16) _Float16 o0[8], o1[8];
      #pragma unroll
      for(int i = 0; i < 8; i++){
        o0[i] = (_Float16)fast_tanh(s0[i] * kScale);
        o1[i] = (_Float16)fast_tanh(s1[i] * kScale);
      }
      *(v8h*)&sS[eL][m*16 + lh*8]    = *(v8h*)o0;
      *(v8h*)&sS[eL+16][m*16 + lh*8] = *(v8h*)o1;
    }
    __syncthreads();
    #pragma unroll
    for(int kc = 0; kc < 2; ++kc){
      v16h vf  = mk16(*(v8h*)&sV[l16][kc*32 + lh*16],   *(v8h*)&sV[l16][kc*32 + lh*16 + 8]);
      v16h sf0 = mk16(*(v8h*)&sS[eL][kc*32 + lh*16],    *(v8h*)&sS[eL][kc*32 + lh*16 + 8]);
      v16h sf1 = mk16(*(v8h*)&sS[eL+16][kc*32 + lh*16], *(v8h*)&sS[eL+16][kc*32 + lh*16 + 8]);
      accw0 = wmma16(vf, sf0, accw0);
      accw1 = wmma16(vf, sf1, accw1);
    }
  }
  #pragma unroll
  for(int s = 0; s < 2; s++){
    int e = e0 + eL + s * 16;
    v8f a = s ? accw1 : accw0;
    __align__(16) _Float16 o[8];
    #pragma unroll
    for(int i = 0; i < 8; i++){
      float w = fast_tanh(a[i]);
      o[i] = (_Float16)(w > 0.f ? w : 0.f);
    }
    *(v8h*)(outp + ((long)(b * Ee + e)) * 16 + lh*8) = *(v8h*)o;
  }
}

// ---------- fusion: res = audio + video + relu(out1@Wf+bf) + relu(out2@Wf+bf) ----------
__global__ void k_fuse(const _Float16* __restrict__ out1, const _Float16* __restrict__ out2,
                       const _Float16* __restrict__ WT, const float* __restrict__ bias,
                       const float* __restrict__ audio, const float* __restrict__ video,
                       float* __restrict__ res){
  __shared__ __align__(16) _Float16 A1[16][64];
  __shared__ __align__(16) _Float16 A2[16][64];
  const int b = blockIdx.y;
  const int n0 = blockIdx.x * 128;
  const int tid = threadIdx.x;
  const int lane = tid & 31, wave = tid >> 5;
  const int l16 = lane & 15, lh = lane >> 4;
  v8f a10 = zero8f(), a11 = zero8f(), a20 = zero8f(), a21 = zero8f();
  const int se = tid & 63, st8 = (tid >> 6) * 8;   // A stage map: e col, 8-t half
  const _Float16* brow = WT + (long)(n0 + wave*32 + l16) * Ee + lh*16;
  for(int kb = 0; kb < Ee; kb += 64){
    __syncthreads();
    {
      v8h x1 = *(const v8h*)(out1 + ((long)(b * Ee + kb + se)) * 16 + st8);
      v8h x2 = *(const v8h*)(out2 + ((long)(b * Ee + kb + se)) * 16 + st8);
      #pragma unroll
      for(int j = 0; j < 8; j++){ A1[st8 + j][se] = x1[j]; A2[st8 + j][se] = x2[j]; }
    }
    __syncthreads();
    __builtin_prefetch(brow + kb + 64, 0, 1);
    __builtin_prefetch(brow + (long)16*Ee + kb + 64, 0, 1);
    #pragma unroll
    for(int kc = 0; kc < 2; kc++){
      v16h af1 = mk16(*(v8h*)&A1[l16][kc*32 + lh*16], *(v8h*)&A1[l16][kc*32 + lh*16 + 8]);
      v16h af2 = mk16(*(v8h*)&A2[l16][kc*32 + lh*16], *(v8h*)&A2[l16][kc*32 + lh*16 + 8]);
      const _Float16* bp  = brow + kb + kc*32;
      const _Float16* bp1 = bp + (long)16 * Ee;
      v16h b0 = mk16(*(const v8h*)bp,  *(const v8h*)(bp  + 8));
      v16h b1 = mk16(*(const v8h*)bp1, *(const v8h*)(bp1 + 8));
      a10 = wmma16(af1, b0, a10);
      a11 = wmma16(af1, b1, a11);
      a20 = wmma16(af2, b0, a20);
      a21 = wmma16(af2, b1, a21);
    }
  }
  #pragma unroll
  for(int s = 0; s < 2; s++){
    int n = n0 + wave*32 + s*16 + l16;
    float bv = bias[n];
    v8f x1 = s ? a11 : a10;
    v8f x2 = s ? a21 : a20;
    #pragma unroll
    for(int i = 0; i < 8; i++){
      int t = i + 8*lh;
      if(t < Tt){
        long idx = ((long)b * Tt + t) * Dd + n;
        float r1 = x1[i] + bv; r1 = r1 > 0.f ? r1 : 0.f;
        float r2 = x2[i] + bv; r2 = r2 > 0.f ? r2 : 0.f;
        res[idx] = audio[idx] + video[idx] + r1 + r2;
      }
    }
  }
}

extern "C" void kernel_launch(void* const* d_in, const int* in_sizes, int n_in,
                              void* d_out, int out_size, void* d_ws, size_t ws_size,
                              hipStream_t stream){
  (void)in_sizes; (void)n_in; (void)out_size; (void)ws_size;
  const float* audio = (const float*)d_in[0];
  const float* video = (const float*)d_in[1];
  const float* Wq  = (const float*)d_in[2];
  const float* bq  = (const float*)d_in[3];
  const float* Wk1 = (const float*)d_in[4];
  const float* bk1 = (const float*)d_in[5];
  const float* Wk2 = (const float*)d_in[6];
  const float* bk2 = (const float*)d_in[7];
  const float* Wv1 = (const float*)d_in[8];
  const float* bv1 = (const float*)d_in[9];
  const float* Wv2 = (const float*)d_in[10];
  const float* bv2 = (const float*)d_in[11];
  const float* Wf  = (const float*)d_in[12];
  const float* bf  = (const float*)d_in[13];
  float* out = (float*)d_out;

  char* ws = (char*)d_ws;
  size_t off = 0;
  auto alloc = [&](size_t bytes) -> _Float16* {
    _Float16* p = (_Float16*)(ws + off);
    off += (bytes + 255) & ~(size_t)255;
    return p;
  };
  _Float16* WqT   = alloc((size_t)Ee * Ee * 2);      // 8 MB
  _Float16* Wv1T  = alloc((size_t)Dd * Dd * 2);      // 2 MB
  _Float16* Wv2T  = alloc((size_t)Dd * Dd * 2);      // 2 MB
  _Float16* WfT   = alloc((size_t)Dd * Ee * 2);      // 4 MB
  _Float16* key1  = alloc((size_t)Bb * Dd * 16 * 2); // 2 MB
  _Float16* key2  = alloc((size_t)Bb * Dd * 16 * 2); // 2 MB
  _Float16* joint = alloc((size_t)Bb * Ee * 16 * 2); // 4 MB
  _Float16* val1  = alloc((size_t)Bb * 16 * Dd * 2); // 2 MB
  _Float16* val2  = alloc((size_t)Bb * 16 * Dd * 2); // 2 MB
  _Float16* o1    = alloc((size_t)Bb * Ee * 16 * 2); // 4 MB
  _Float16* o2    = alloc((size_t)Bb * Ee * 16 * 2); // 4 MB  (~36 MB total)

  // weights -> f16 transposed [N][K] (LDS-tiled, coalesced both sides)
  hipLaunchKernelGGL(k_wtrans, dim3(Ee/32, Ee/32), dim3(32, 8), 0, stream, Wq,  WqT,  Ee, Ee);
  hipLaunchKernelGGL(k_wtrans, dim3(Dd/32, Dd/32), dim3(32, 8), 0, stream, Wv1, Wv1T, Dd, Dd);
  hipLaunchKernelGGL(k_wtrans, dim3(Dd/32, Dd/32), dim3(32, 8), 0, stream, Wv2, Wv2T, Dd, Dd);
  hipLaunchKernelGGL(k_wtrans, dim3(Ee/32, Dd/32), dim3(32, 8), 0, stream, Wf,  WfT,  Ee, Dd);
  // tiny T x T key projections (K=10, not WMMA-worthy)
  hipLaunchKernelGGL(k_keyproj, dim3((Bb * Dd) / 256), dim3(256), 0, stream, audio, Wk1, bk1, key1);
  hipLaunchKernelGGL(k_keyproj, dim3((Bb * Dd) / 256), dim3(256), 0, stream, video, Wk2, bk2, key2);
  // WMMA projections
  hipLaunchKernelGGL(k_projjoint, dim3(Ee / 128, Bb), dim3(128), 0, stream, audio, video, WqT, bq, joint);
  hipLaunchKernelGGL(k_projval,   dim3(Dd / 128, Bb), dim3(128), 0, stream, audio, Wv1T, bv1, val1);
  hipLaunchKernelGGL(k_projval,   dim3(Dd / 128, Bb), dim3(128), 0, stream, video, Wv2T, bv2, val2);
  // fused attention branches (536 MB/branch scores tensor stays in LDS/registers)
  hipLaunchKernelGGL(k_attn, dim3(Ee / 128, Bb), dim3(128), 0, stream, key1, val1, joint, o1);
  hipLaunchKernelGGL(k_attn, dim3(Ee / 128, Bb), dim3(128), 0, stream, key2, val2, joint, o2);
  // fusion GEMM + residual
  hipLaunchKernelGGL(k_fuse, dim3(Dd / 128, Bb), dim3(128), 0, stream, o1, o2, WfT, bf, audio, video, out);
}
